// CMultiHeadAttention_69965017252615
// MI455X (gfx1250) — matrix operand order repaired and checked by reading, not compiled, not run on hardware
//
#include <hip/hip_runtime.h>
#include <hip/hip_bf16.h>
#include <stdint.h>
#include <stddef.h>

#define Bdim  4
#define Ldim  1024
#define Ddim  1024
#define Hdim  16
#define DHdim 64

typedef __attribute__((ext_vector_type(16))) __bf16 v16bf;
typedef __attribute__((ext_vector_type(8)))  float  v8f;

__device__ __forceinline__ unsigned int pk_bf16(float a, float b) {
  union { __bf16 h[2]; unsigned int u; } un;
  un.h[0] = (__bf16)a;
  un.h[1] = (__bf16)b;
  return un.u;
}

// Fragment-order LDS tile: tile[lane][16] bf16 (32B/lane, 1KB per 16x32 tile).
// Loading a fragment = one contiguous 32B read per lane (2x ds_load_b128).
__device__ __forceinline__ v16bf load_frag(const __bf16* tile) {
  return *(const v16bf*)(tile + (threadIdx.x & 31) * 16);
}

// A-tile (16M x 32K) element (m,k) lives at lane=(m&15)+16*((k>>3)&1),
//   f = (k&16 ? 8 : 0) + (k&7)         [from ISA 7.12.2 16-bit A layout]
// B-tile (32K x 16N) element (k,n) lives at lane=(n&15)+16*((k>>3)&1), same f as A.

__device__ __forceinline__ v8f wmma_bf16(v16bf a, v16bf b, v8f c) {
  return __builtin_amdgcn_wmma_f32_16x16x32_bf16(false, a, false, b,
                                                 (short)0, c, false, false);
}

__device__ __forceinline__ float softplusf(float x) {
  return (x > 15.f) ? x : log1pf(__expf(x));
}

// Async global->LDS 16B copy (per-lane LDS dest addressing), ASYNCcnt-tracked.
// Low 32 bits of a generic LDS pointer are the wave-relative LDS byte offset.
__device__ __forceinline__ void async_copy_b128(void* lds, const void* gptr) {
  const unsigned lds_off = (unsigned)(uintptr_t)lds;
  asm volatile("global_load_async_to_lds_b128 %0, %1, off"
               :
               : "v"(lds_off), "v"(gptr)
               : "memory");
}

__device__ __forceinline__ void wait_async0() {
#if __has_builtin(__builtin_amdgcn_s_wait_asynccnt)
  __builtin_amdgcn_s_wait_asynccnt(0);
#else
  asm volatile("s_wait_asynccnt 0x0" ::: "memory");
#endif
}

// ---------------------------------------------------------------------------
// Y = X[M,K] * W[N,K]^T (+bias). Block tile 128x64, 8 waves, wave tile 32x32.
// MODE 0: bf16 out head-split [B,H,L,DH]
// MODE 1: f32 out flat [M,N]
// MODE 2: bf16 out transposed head-split [B,H,DH,L]  (for V -> attn_v B-side)
// ---------------------------------------------------------------------------
template <int MODE>
__global__ __launch_bounds__(256) void gemm_xwT(
    const float* __restrict__ X, const float* __restrict__ W,
    const float* __restrict__ bias, __bf16* __restrict__ Ybf,
    float* __restrict__ Yf, int M, int N, int K) {
  __shared__ __attribute__((aligned(32))) __bf16 Af[8][512];  // 8 M-tiles
  __shared__ __attribute__((aligned(32))) __bf16 Bf[4][512];  // 4 N-tiles
  const int tid = threadIdx.x, lane = tid & 31, warp = tid >> 5;
  const int wm = warp >> 1, wn = warp & 1;
  const int m0 = blockIdx.y * 128, n0 = blockIdx.x * 64;

  v8f acc[2][2];
#pragma unroll
  for (int i = 0; i < 2; ++i)
#pragma unroll
    for (int j = 0; j < 2; ++j)
#pragma unroll
      for (int e = 0; e < 8; ++e) acc[i][j][e] = 0.f;

  for (int k0 = 0; k0 < K; k0 += 32) {
    __syncthreads();
    // Stage A: 128x32 f32 -> fragment-order bf16 (1024 float4 loads)
#pragma unroll
    for (int j = 0; j < 4; ++j) {
      const int v = j * 256 + tid;
      const int row = v >> 3, c = (v & 7) << 2;  // c in {0,4,...,28}
      float4 x = *(const float4*)(X + (size_t)(m0 + row) * K + k0 + c);
      const int mt = row >> 4;
      const int ln = (row & 15) + (((c >> 3) & 1) << 4);
      const int f0 = ((c & 16) >> 1) + (c & 7);  // {0,4,8,12}
      uint2 p;
      p.x = pk_bf16(x.x, x.y);
      p.y = pk_bf16(x.z, x.w);
      *(uint2*)&Af[mt][ln * 16 + f0] = p;
    }
    // Stage B: W[n,k] (coalesced along K) -> fragment order (512 float4)
#pragma unroll
    for (int j = 0; j < 2; ++j) {
      const int v = j * 256 + tid;
      const int n = v >> 3, c = (v & 7) << 2;
      float4 x = *(const float4*)(W + (size_t)(n0 + n) * K + k0 + c);
      const int nt = n >> 4;
      const int ln = (n & 15) + (((c >> 3) & 1) << 4);
      const int f0 = ((c & 16) >> 1) + (c & 7);  // {0,4,8,12}
      uint2 p;
      p.x = pk_bf16(x.x, x.y);
      p.y = pk_bf16(x.z, x.w);
      *(uint2*)&Bf[nt][ln * 16 + f0] = p;
    }
    __syncthreads();

    v16bf afr[2], bfr[2];
#pragma unroll
    for (int tm = 0; tm < 2; ++tm) afr[tm] = load_frag(Af[wm * 2 + tm]);
#pragma unroll
    for (int tn = 0; tn < 2; ++tn) bfr[tn] = load_frag(Bf[wn * 2 + tn]);
#pragma unroll
    for (int tm = 0; tm < 2; ++tm)
#pragma unroll
      for (int tn = 0; tn < 2; ++tn)
        acc[tm][tn] = wmma_bf16(afr[tm], bfr[tn], acc[tm][tn]);
  }

  const int rofs = 8 * (lane >> 4), cofs = lane & 15;
#pragma unroll
  for (int tm = 0; tm < 2; ++tm)
#pragma unroll
    for (int tn = 0; tn < 2; ++tn)
#pragma unroll
      for (int r = 0; r < 8; ++r) {
        const int m = m0 + wm * 32 + tm * 16 + r + rofs;
        const int n = n0 + wn * 32 + tn * 16 + cofs;
        float val = acc[tm][tn][r];
        if (bias) val += bias[n];
        const int b = m >> 10, l = m & 1023;
        const int h = n >> 6, d = n & 63;
        if (MODE == 0) {
          Ybf[(((size_t)b * Hdim + h) * Ldim + l) * DHdim + d] = (__bf16)val;
        } else if (MODE == 2) {
          Ybf[(((size_t)b * Hdim + h) * DHdim + d) * Ldim + l] = (__bf16)val;
        } else {
          Yf[(size_t)m * N + n] = val;
        }
      }
}

// ---------------------------------------------------------------------------
// Scores + relative bias. Grid (L/128 k, L/128 q, B*H). Block 256 = 8 waves,
// wave tile 32q x 64k. DH=64 staged once (async copies) as 2x 32-deep sets.
// ---------------------------------------------------------------------------
__global__ __launch_bounds__(256) void scores_bias(
    const __bf16* __restrict__ qh, const __bf16* __restrict__ kh,
    const float* __restrict__ d_q, const float* __restrict__ d_k_top,
    const float* __restrict__ d_k_bot, const float* __restrict__ d_k_score,
    const float* __restrict__ rel_top, const float* __restrict__ rel_bot,
    const float* __restrict__ W_w, const float* __restrict__ W_b,
    float* __restrict__ attn) {
  __shared__ __attribute__((aligned(32))) __bf16 Af[2][8][512];  // [kk][qtile]
  __shared__ __attribute__((aligned(32))) __bf16 Bf[2][8][512];  // [kk][ktile]
  __shared__ float dqs[128][4];
  __shared__ float kf[128][20];  // [0:3)=dkt [3:6)=dkb [6:13)=s_top [13:20)=s_bot
  __shared__ float wWs[8], wBs[8];

  const int tid = threadIdx.x, lane = tid & 31, warp = tid >> 5;
  const int wq = warp >> 1, wk = warp & 1;
  const int k0 = blockIdx.x * 128, q0 = blockIdx.y * 128;
  const int bh = blockIdx.z;
  const int b = bh >> 4, h = bh & 15;

  const __bf16* qbase = qh + ((size_t)bh * Ldim + q0) * DHdim;
  const __bf16* kbase = kh + ((size_t)bh * Ldim + k0) * DHdim;

  // Q tile 128x64: async copy straight into A-fragment order (8 dh per b128)
#pragma unroll
  for (int j = 0; j < 4; ++j) {
    const int v = j * 256 + tid;
    const int row = v >> 3, c = (v & 7) * 8;  // c in {0,8,...,56}
    const int ks = c >> 5, mt = row >> 4;
    const int ln = (row & 15) + (((c >> 3) & 1) << 4);
    const int f0 = (c & 16) >> 1;  // {0,8}
    async_copy_b128(&Af[ks][mt][ln * 16 + f0], qbase + (size_t)row * DHdim + c);
  }
  // K tile 128x64: async copy into B-fragment order
#pragma unroll
  for (int j = 0; j < 4; ++j) {
    const int v = j * 256 + tid;
    const int n = v >> 3, c = (v & 7) * 8;
    const int ks = c >> 5, nt = n >> 4;
    const int ln = (n & 15) + (((c >> 3) & 1) << 4);
    const int f0 = (c & 16) >> 1;  // {0,8}
    async_copy_b128(&Bf[ks][nt][ln * 16 + f0], kbase + (size_t)n * DHdim + c);
  }
  if (tid < 128) {
    const int q = q0 + tid;
#pragma unroll
    for (int f = 0; f < 3; ++f) dqs[tid][f] = d_q[((size_t)b * Ldim + q) * 3 + f];
    const int k = k0 + tid;
#pragma unroll
    for (int f = 0; f < 3; ++f) {
      kf[tid][f]     = d_k_top[((size_t)b * Ldim + k) * 3 + f];
      kf[tid][3 + f] = d_k_bot[((size_t)b * Ldim + k) * 3 + f];
    }
#pragma unroll
    for (int f = 0; f < 7; ++f) {
      kf[tid][6 + f]  = d_k_score[(((size_t)b * Ldim + k) * 2 + 0) * 7 + f];
      kf[tid][13 + f] = d_k_score[(((size_t)b * Ldim + k) * 2 + 1) * 7 + f];
    }
  }
  if (tid < 7) {
    wWs[tid] = W_w[h * 7 + tid];
    wBs[tid] = W_b[h * 7 + tid];
  }
  wait_async0();
  __syncthreads();

  v8f acc[2][4];
#pragma unroll
  for (int i = 0; i < 2; ++i)
#pragma unroll
    for (int j = 0; j < 4; ++j)
#pragma unroll
      for (int e = 0; e < 8; ++e) acc[i][j][e] = 0.f;

#pragma unroll
  for (int ks = 0; ks < 2; ++ks) {
    v16bf afr[2], bfr[4];
#pragma unroll
    for (int tm = 0; tm < 2; ++tm) afr[tm] = load_frag(Af[ks][wq * 2 + tm]);
#pragma unroll
    for (int tn = 0; tn < 4; ++tn) bfr[tn] = load_frag(Bf[ks][wk * 4 + tn]);
#pragma unroll
    for (int tm = 0; tm < 2; ++tm)
#pragma unroll
      for (int tn = 0; tn < 4; ++tn)
        acc[tm][tn] = wmma_bf16(afr[tm], bfr[tn], acc[tm][tn]);
  }

  const float* rtb = rel_top + (size_t)b * Ldim * Ldim * 4;
  const float* rbb = rel_bot + (size_t)b * Ldim * Ldim * 4;
  float* ab = attn + (size_t)bh * Ldim * Ldim;
  const int rofs = 8 * (lane >> 4), cofs = lane & 15;
  float w7[7], b7[7];
#pragma unroll
  for (int f = 0; f < 7; ++f) { w7[f] = wWs[f]; b7[f] = wBs[f]; }

  for (int tm = 0; tm < 2; ++tm)
    for (int tn = 0; tn < 4; ++tn)
      for (int r = 0; r < 8; ++r) {
        const int lq = wq * 32 + tm * 16 + r + rofs;
        const int lk = wk * 64 + tn * 16 + cofs;
        const int q = q0 + lq, k = k0 + lk;
        float4 rt = *(const float4*)(rtb + ((size_t)q * Ldim + k) * 4);
        float4 rb = *(const float4*)(rbb + ((size_t)q * Ldim + k) * 4);
        float dist[7];
#pragma unroll
        for (int f = 0; f < 3; ++f)
          dist[f] = (dqs[lq][f] - kf[lk][f]) * kf[lk][6 + f] +
                    (dqs[lq][f] - kf[lk][3 + f]) * kf[lk][13 + f];
        dist[3] = rt.x * kf[lk][9]  + rb.x * kf[lk][16];
        dist[4] = rt.y * kf[lk][10] + rb.y * kf[lk][17];
        dist[5] = rt.z * kf[lk][11] + rb.z * kf[lk][18];
        dist[6] = rt.w * kf[lk][12] + rb.w * kf[lk][19];
        float ww = 0.f, bb = 0.f;
#pragma unroll
        for (int f = 0; f < 7; ++f) {
          ww += dist[f] * w7[f];
          bb += dist[f] * b7[f];
        }
        const float sc = acc[tm][tn][r] * 0.125f * softplusf(ww) + bb;
        ab[(size_t)q * Ldim + k] = sc;
      }
}

// ---------------------------------------------------------------------------
// In-place row softmax over the last dim (L=1024). One block per row.
// ---------------------------------------------------------------------------
__global__ __launch_bounds__(256) void softmax_rows(float* __restrict__ attn) {
  __shared__ float redm[8], reds[8];
  float* p = attn + (size_t)blockIdx.x * Ldim;
  const int tid = threadIdx.x;
  float4 x = *(float4*)(p + tid * 4);
  float m = fmaxf(fmaxf(x.x, x.y), fmaxf(x.z, x.w));
#pragma unroll
  for (int o = 16; o > 0; o >>= 1) m = fmaxf(m, __shfl_xor(m, o));
  if ((tid & 31) == 0) redm[tid >> 5] = m;
  __syncthreads();
  float bm = redm[0];
#pragma unroll
  for (int i = 1; i < 8; ++i) bm = fmaxf(bm, redm[i]);
  x.x = __expf(x.x - bm);
  x.y = __expf(x.y - bm);
  x.z = __expf(x.z - bm);
  x.w = __expf(x.w - bm);
  float s = x.x + x.y + x.z + x.w;
#pragma unroll
  for (int o = 16; o > 0; o >>= 1) s += __shfl_xor(s, o);
  if ((tid & 31) == 0) reds[tid >> 5] = s;
  __syncthreads();
  float ts = 0.f;
#pragma unroll
  for (int i = 0; i < 8; ++i) ts += reds[i];
  const float inv = 1.0f / ts;
  x.x *= inv; x.y *= inv; x.z *= inv; x.w *= inv;
  *(float4*)(p + tid * 4) = x;
}

// ---------------------------------------------------------------------------
// ctx[b,q,h*64+d] = sum_k attn[b,h,q,k] * v[b,h,k,d], V given transposed
// as vT[b,h,d,l] so B-side staging is contiguous-in-K (async copies).
// ---------------------------------------------------------------------------
__global__ __launch_bounds__(256) void attn_v(
    const float* __restrict__ attn, const __bf16* __restrict__ vT,
    float* __restrict__ ctx) {
  __shared__ __attribute__((aligned(32))) __bf16 Af[8][512];
  __shared__ __attribute__((aligned(32))) __bf16 Bf[4][512];
  const int tid = threadIdx.x, lane = tid & 31, warp = tid >> 5;
  const int wm = warp >> 1, wn = warp & 1;
  const int q0 = blockIdx.x * 128;
  const int bh = blockIdx.y;
  const int b = bh >> 4, h = bh & 15;
  const float* ab = attn + (size_t)bh * Ldim * Ldim;
  const __bf16* vTb = vT + (size_t)bh * DHdim * Ldim;

  v8f acc[2][2];
#pragma unroll
  for (int i = 0; i < 2; ++i)
#pragma unroll
    for (int j = 0; j < 2; ++j)
#pragma unroll
      for (int e = 0; e < 8; ++e) acc[i][j][e] = 0.f;

  for (int k0 = 0; k0 < Ldim; k0 += 32) {
    __syncthreads();
    // B: vT rows contiguous in K: async straight into fragment order
    {
      const int n = tid >> 2, c = (tid & 3) * 8;
      const int nt = n >> 4;
      const int ln = (n & 15) + (((c >> 3) & 1) << 4);
      const int f0 = (c & 16) >> 1;  // {0,8}
      async_copy_b128(&Bf[nt][ln * 16 + f0], vTb + (size_t)n * Ldim + k0 + c);
    }
    // A: attn f32 128x32 -> convert to fragment-order bf16
#pragma unroll
    for (int j = 0; j < 4; ++j) {
      const int v = j * 256 + tid;
      const int row = v >> 3, c = (v & 7) << 2;
      float4 x = *(const float4*)(ab + (size_t)(q0 + row) * Ldim + k0 + c);
      const int mt = row >> 4;
      const int ln = (row & 15) + (((c >> 3) & 1) << 4);
      const int f0 = ((c & 16) >> 1) + (c & 7);
      uint2 p;
      p.x = pk_bf16(x.x, x.y);
      p.y = pk_bf16(x.z, x.w);
      *(uint2*)&Af[mt][ln * 16 + f0] = p;
    }
    wait_async0();
    __syncthreads();

    v16bf afr[2], bfr[2];
#pragma unroll
    for (int tm = 0; tm < 2; ++tm) afr[tm] = load_frag(Af[wm * 2 + tm]);
#pragma unroll
    for (int tn = 0; tn < 2; ++tn) bfr[tn] = load_frag(Bf[wn * 2 + tn]);
#pragma unroll
    for (int tm = 0; tm < 2; ++tm)
#pragma unroll
      for (int tn = 0; tn < 2; ++tn)
        acc[tm][tn] = wmma_bf16(afr[tm], bfr[tn], acc[tm][tn]);
  }

  const int rofs = 8 * (lane >> 4), cofs = lane & 15;
#pragma unroll
  for (int tm = 0; tm < 2; ++tm)
#pragma unroll
    for (int tn = 0; tn < 2; ++tn)
#pragma unroll
      for (int r = 0; r < 8; ++r) {
        const int m = q0 + wm * 32 + tm * 16 + r + rofs;
        const int n = wn * 32 + tn * 16 + cofs;
        ctx[((size_t)b * Ldim + m) * Ddim + h * 64 + n] = acc[tm][tn][r];
      }
}

// ---------------------------------------------------------------------------
extern "C" void kernel_launch(void* const* d_in, const int* in_sizes, int n_in,
                              void* d_out, int out_size, void* d_ws,
                              size_t ws_size, hipStream_t stream) {
  (void)in_sizes; (void)n_in; (void)out_size; (void)ws_size;
  const float* q      = (const float*)d_in[0];
  const float* k      = (const float*)d_in[1];
  const float* v      = (const float*)d_in[2];
  const float* d_q    = (const float*)d_in[3];
  const float* d_ktop = (const float*)d_in[4];
  const float* d_kbot = (const float*)d_in[5];
  const float* d_ks   = (const float*)d_in[6];
  const float* r_top  = (const float*)d_in[7];
  const float* r_bot  = (const float*)d_in[8];
  const float* W_q    = (const float*)d_in[9];
  const float* b_q    = (const float*)d_in[10];
  const float* W_k    = (const float*)d_in[11];
  const float* W_v    = (const float*)d_in[12];
  // d_in[13] = W_c: its output `c` is unused by the reference outputs.
  const float* W_w    = (const float*)d_in[14];
  const float* W_b    = (const float*)d_in[15];
  const float* W_o    = (const float*)d_in[16];
  const float* b_o    = (const float*)d_in[17];

  const size_t nBLD = (size_t)Bdim * Ldim * Ddim;  // 4,194,304
  __bf16* qh  = (__bf16*)d_ws;
  __bf16* kh  = qh + nBLD;
  __bf16* vhT = kh + nBLD;  // [B,H,DH,L]
  float* ctx  = (float*)(vhT + nBLD);

  float* out  = (float*)d_out;
  float* attn = out + nBLD;  // [B,H,L,L]

  dim3 gproj(Ddim / 64, (Bdim * Ldim) / 128);  // (16, 32)
  gemm_xwT<0><<<gproj, 256, 0, stream>>>(q, W_q, b_q, qh, nullptr,
                                         Bdim * Ldim, Ddim, Ddim);
  gemm_xwT<0><<<gproj, 256, 0, stream>>>(k, W_k, nullptr, kh, nullptr,
                                         Bdim * Ldim, Ddim, Ddim);
  gemm_xwT<2><<<gproj, 256, 0, stream>>>(v, W_v, nullptr, vhT, nullptr,
                                         Bdim * Ldim, Ddim, Ddim);

  dim3 gsc(Ldim / 128, Ldim / 128, Bdim * Hdim);  // (8, 8, 64)
  scores_bias<<<gsc, 256, 0, stream>>>(qh, kh, d_q, d_ktop, d_kbot, d_ks,
                                       r_top, r_bot, W_w, W_b, attn);

  softmax_rows<<<Bdim * Hdim * Ldim, 256, 0, stream>>>(attn);

  dim3 gav(Ldim / 128, Bdim * Hdim);  // (8, 64)
  attn_v<<<gav, 256, 0, stream>>>(attn, vhT, ctx);

  gemm_xwT<1><<<gproj, 256, 0, stream>>>(ctx, W_o, b_o, nullptr, out,
                                         Bdim * Ldim, Ddim, Ddim);
}
